// CrystalDynamics_40965398069689
// MI455X (gfx1250) — matrix operand order repaired and checked by reading, not compiled, not run on hardware
//
#include <hip/hip_runtime.h>
#include <hip/hip_bf16.h>
#include <math.h>

typedef __attribute__((ext_vector_type(16))) _Float16 v16h;
typedef __attribute__((ext_vector_type(8)))  _Float16 v8h;
typedef __attribute__((ext_vector_type(4)))  _Float16 v4h;
typedef __attribute__((ext_vector_type(8)))  float    v8f;

#define NN     8192
#define BB     64
#define NODE   64
#define TIMEDIM 64
#define ZDIM   128
#define KNB    12
#define NLAYER 4
#define NEDGE  (NN*KNB)   // 98304

__device__ __forceinline__ float silu_f(float x) {
    return x / (1.0f + __expf(-x));
}

// A-fragment loader: 16x32 f16 A in this code's contiguous k order (matches pack_weights). Row pointer = start of row m in LDS
// (f16, row-major). Per lane (half = lane>>4): halves 0..7  = K kbase+half*16..+7,
// halves 8..15 = K kbase+half*16+8..+15.
__device__ __forceinline__ v16h load_a_frag(const _Float16* rowp, int kbase, int half) {
    v8h lo = *(const v8h*)(rowp + kbase + half * 16);
    v8h hi = *(const v8h*)(rowp + kbase + half * 16 + 8);
    v16h a;
#pragma unroll
    for (int i = 0; i < 8; ++i) { a[i] = lo[i]; a[i + 8] = hi[i]; }
    return a;
}

__device__ __forceinline__ v8f wmma_f16f32(v16h a, v16h b, v8f c) {
    return __builtin_amdgcn_wmma_f32_16x16x32_f16(false, a, false, b, (short)0, c, false, false);
}

__device__ __forceinline__ void zero_acc4(v8f* acc) {
#pragma unroll
    for (int nt = 0; nt < 4; ++nt)
#pragma unroll
        for (int r = 0; r < 8; ++r) acc[nt][r] = 0.0f;
}

__device__ __forceinline__ v4h cvt4(float4 v) {
    v4h h;
    h[0] = (_Float16)v.x; h[1] = (_Float16)v.y;
    h[2] = (_Float16)v.z; h[3] = (_Float16)v.w;
    return h;
}

// ---------------------------------------------------------------------------
// Utility kernels
// ---------------------------------------------------------------------------
__global__ void zero_f32_kernel(float* p, int n) {
    int i = blockIdx.x * blockDim.x + threadIdx.x;
    if (i < n) p[i] = 0.0f;
}

// Pack fp32 weight matrix W[K][64] (row-major) into f16 B-fragment order,
// zero-padded to ksteps*32 rows:
//   tile = ks*4+nt ; per lane: n = nt*16+(lane&15), half = lane>>4
//   half element 2p+q  <->  k = ks*32 + half*16 + 2p + q
// edge_perm!=0 applies the edge-input column permutation:
//   staged layout [h_src|h_dst (0..127) | t(128..191) | dsq(192)]
//   maps to reference rows: k<128 -> k ; 128<=k<192 -> k+1 ; k==192 -> 128
__global__ void pack_weights_kernel(const float* __restrict__ W, _Float16* __restrict__ out,
                                    int K, int ksteps, int edge_perm) {
    int idx  = blockIdx.x * blockDim.x + threadIdx.x;
    int lane = idx & 31;
    int tile = idx >> 5;
    if (tile >= ksteps * 4) return;
    int ks = tile >> 2, nt = tile & 3;
    int n    = nt * 16 + (lane & 15);
    int half = (lane >> 4) & 1;
    _Float16* dst = out + (size_t)tile * 512 + (size_t)lane * 16;
#pragma unroll
    for (int p = 0; p < 8; ++p) {
#pragma unroll
        for (int q = 0; q < 2; ++q) {
            int k = ks * 32 + half * 16 + 2 * p + q;
            int ksrc;
            if (edge_perm) {
                if (k < 128)       ksrc = k;
                else if (k < 192)  ksrc = k + 1;
                else if (k == 192) ksrc = 128;
                else               ksrc = -1;
            } else {
                ksrc = (k < K) ? k : -1;
            }
            float v = (ksrc >= 0 && ksrc < K) ? W[(size_t)ksrc * 64 + n] : 0.0f;
            dst[p * 2 + q] = (_Float16)v;
        }
    }
}

// ---------------------------------------------------------------------------
// Time-embedding MLP: te[b] = silu(sin_emb(t[b]) @ W1 + b1) @ W2 + b2
// grid=64 (one block per batch), block=64
// ---------------------------------------------------------------------------
__global__ void time_mlp_kernel(const float* __restrict__ t,
                                const float* __restrict__ w1, const float* __restrict__ b1,
                                const float* __restrict__ w2, const float* __restrict__ b2,
                                float* __restrict__ te) {
    __shared__ float s[64];
    __shared__ float hid[128];
    int b = blockIdx.x;
    int j = threadIdx.x;  // 0..63
    float tv = t[b];
    {
        int i = j & 31;
        float freq = __expf((float)i * (-logf(10000.0f) / 31.0f));
        float ang = tv * freq;
        s[j] = (j < 32) ? sinf(ang) : cosf(ang);
    }
    __syncthreads();
    for (int c = j; c < 128; c += 64) {
        float acc = b1[c];
#pragma unroll 1
        for (int k = 0; k < 64; ++k) acc += s[k] * w1[k * 128 + c];
        hid[c] = silu_f(acc);
    }
    __syncthreads();
    float acc = b2[j];
#pragma unroll 1
    for (int k = 0; k < 128; ++k) acc += hid[k] * w2[k * 64 + j];
    te[b * 64 + j] = acc;
}

// ---------------------------------------------------------------------------
// Brute-force 12-NN per graph (128 nodes). grid=64, block=128.
// ---------------------------------------------------------------------------
__global__ void knn_kernel(const float* __restrict__ cart, int* __restrict__ edge_src) {
    __shared__ float cx[128], cy[128], cz[128];
    int g = blockIdx.x;
    int i = threadIdx.x;
    int gi = g * 128 + i;
    cx[i] = cart[gi * 3 + 0];
    cy[i] = cart[gi * 3 + 1];
    cz[i] = cart[gi * 3 + 2];
    __syncthreads();
    float xi = cx[i], yi = cy[i], zi = cz[i];
    unsigned mask[4] = {0u, 0u, 0u, 0u};
    mask[i >> 5] |= 1u << (i & 31);  // exclude self
#pragma unroll 1
    for (int k = 0; k < KNB; ++k) {
        float best = 1e30f;
        int bj = 0;
#pragma unroll 1
        for (int j = 0; j < 128; ++j) {
            if ((mask[j >> 5] >> (j & 31)) & 1u) continue;
            float dx = cx[j] - xi, dy = cy[j] - yi, dz = cz[j] - zi;
            float d = dx * dx + dy * dy + dz * dz;
            if (d < best) { best = d; bj = j; }
        }
        mask[bj >> 5] |= 1u << (bj & 31);
        edge_src[gi * KNB + k] = g * 128 + bj;
    }
}

// ---------------------------------------------------------------------------
// Embedding GEMM: h = z[8192,128] @ emb_w[128,64] + emb_b  (WMMA f16)
// grid=64, block=256 (8 waves x 16 rows)
// ---------------------------------------------------------------------------
__global__ void embed_kernel(const float* __restrict__ z, const _Float16* __restrict__ pw,
                             const float* __restrict__ bias, float* __restrict__ h) {
    __shared__ alignas(32) _Float16 sm[8 * 16 * 128];
    int lane = threadIdx.x & 31, wave = threadIdx.x >> 5;
    int r0 = (blockIdx.x * 8 + wave) * 16;
    _Float16* A = sm + (size_t)wave * 2048;
    {
        int row = lane >> 1, sub = lane & 1;
        const float* src = z + (size_t)(r0 + row) * 128 + sub * 64;
        _Float16* dstp = A + row * 128 + sub * 64;
#pragma unroll
        for (int j = 0; j < 16; ++j) {
            int cb = j * 4;
            *(v4h*)(dstp + cb) = cvt4(*(const float4*)(src + cb));
        }
    }
    __syncthreads();
    int m = lane & 15, half = lane >> 4;
    v8f acc[4];
    zero_acc4(acc);
    const _Float16* Ab = A + m * 128;
#pragma unroll
    for (int ks = 0; ks < 4; ++ks) {
        v16h a = load_a_frag(Ab, ks * 32, half);
#pragma unroll
        for (int nt = 0; nt < 4; ++nt) {
            v16h b = *(const v16h*)(pw + ((size_t)(ks * 4 + nt) * 32 + lane) * 16);
            acc[nt] = wmma_f16f32(a, b, acc[nt]);
        }
    }
#pragma unroll
    for (int nt = 0; nt < 4; ++nt) {
        int c = nt * 16 + m;
        float bi = bias[c];
#pragma unroll
        for (int r = 0; r < 8; ++r) {
            int row = r0 + r + 8 * half;
            h[(size_t)row * 64 + c] = acc[nt][r] + bi;
        }
    }
}

// ---------------------------------------------------------------------------
// Fused per-layer edge pipeline. grid=768, block=256 (8 waves x 16 edges).
// LDS region per wave: 3584 halves. [0,3584): e_in staging (16x224), permuted
// layout [h_src(64) | h_dst(64) | t_edge(64) | d^2(1) | pad(31)].
// After GEMM1 reads finish: [0,1024)=hid1, [1024,2048)=m, [2048,3072)=ch.
// ---------------------------------------------------------------------------
__global__ void edge_kernel(const float* __restrict__ h, const float* __restrict__ te,
                            const int* __restrict__ batch, const float* __restrict__ cart,
                            const int* __restrict__ edge_src,
                            const _Float16* __restrict__ pw1, const float* __restrict__ eb1,
                            const _Float16* __restrict__ pw2, const float* __restrict__ eb2,
                            const _Float16* __restrict__ pcw1, const float* __restrict__ cb1,
                            const float* __restrict__ cw2,
                            float* __restrict__ m_i, float* __restrict__ coord_out) {
    __shared__ alignas(32) _Float16 sm[8 * 3584];
    int lane = threadIdx.x & 31, wave = threadIdx.x >> 5;
    int e0 = (blockIdx.x * 8 + wave) * 16;
    _Float16* R = sm + (size_t)wave * 3584;

    // ---- stage e_in: branch-free per-segment loops, float4 -> v4h
    {
        int row = lane >> 1, sub = lane & 1;
        int e = e0 + row;
        int src = edge_src[e], dst = e / KNB;
        const float* hs = h + (size_t)src * 64 + sub * 32;
        const float* hd = h + (size_t)dst * 64 + sub * 32;
        const float* tn = te + (size_t)batch[src] * 64 + sub * 32;
        float dx = cart[src * 3 + 0] - cart[dst * 3 + 0];
        float dy = cart[src * 3 + 1] - cart[dst * 3 + 1];
        float dz = cart[src * 3 + 2] - cart[dst * 3 + 2];
        float dsq = dx * dx + dy * dy + dz * dz;
        _Float16* out = R + row * 224 + sub * 32;
#pragma unroll
        for (int j = 0; j < 8; ++j)
            *(v4h*)(out + j * 4) = cvt4(*(const float4*)(hs + j * 4));
#pragma unroll
        for (int j = 0; j < 8; ++j)
            *(v4h*)(out + 64 + j * 4) = cvt4(*(const float4*)(hd + j * 4));
#pragma unroll
        for (int j = 0; j < 8; ++j)
            *(v4h*)(out + 128 + j * 4) = cvt4(*(const float4*)(tn + j * 4));
        // tail cols [192,224): dsq then zero pad; 4 chunks of 4 per sub-lane
        _Float16* tail = R + row * 224 + 192 + sub * 16;
#pragma unroll
        for (int j = 0; j < 4; ++j) {
            v4h zv;
            zv[0] = (_Float16)0.0f; zv[1] = (_Float16)0.0f;
            zv[2] = (_Float16)0.0f; zv[3] = (_Float16)0.0f;
            if (sub == 0 && j == 0) zv[0] = (_Float16)dsq;
            *(v4h*)(tail + j * 4) = zv;
        }
    }
    __syncthreads();

    int m = lane & 15, half = lane >> 4;
    v8f acc[4];

    // ---- GEMM1: e_in[16x224] @ edge_w1(perm) -> hid1, silu
    zero_acc4(acc);
    {
        const _Float16* Ab = R + m * 224;
#pragma unroll
        for (int ks = 0; ks < 7; ++ks) {
            v16h a = load_a_frag(Ab, ks * 32, half);
#pragma unroll
            for (int nt = 0; nt < 4; ++nt) {
                v16h b = *(const v16h*)(pw1 + ((size_t)(ks * 4 + nt) * 32 + lane) * 16);
                acc[nt] = wmma_f16f32(a, b, acc[nt]);
            }
        }
    }
    __syncthreads();
#pragma unroll
    for (int nt = 0; nt < 4; ++nt) {
        int c = nt * 16 + m;
        float bi = eb1[c];
#pragma unroll
        for (int r = 0; r < 8; ++r) {
            float v = silu_f(acc[nt][r] + bi);
            R[(r + 8 * half) * 64 + c] = (_Float16)v;
        }
    }
    __syncthreads();

    // ---- GEMM2: hid1 @ edge_w2 -> m (silu), store LDS + atomic m_i
    zero_acc4(acc);
    {
        const _Float16* Ab = R + m * 64;
#pragma unroll
        for (int ks = 0; ks < 2; ++ks) {
            v16h a = load_a_frag(Ab, ks * 32, half);
#pragma unroll
            for (int nt = 0; nt < 4; ++nt) {
                v16h b = *(const v16h*)(pw2 + ((size_t)(ks * 4 + nt) * 32 + lane) * 16);
                acc[nt] = wmma_f16f32(a, b, acc[nt]);
            }
        }
    }
    __syncthreads();
#pragma unroll
    for (int nt = 0; nt < 4; ++nt) {
        int c = nt * 16 + m;
        float bi = eb2[c];
#pragma unroll
        for (int r = 0; r < 8; ++r) {
            int M = r + 8 * half;
            float v = silu_f(acc[nt][r] + bi);
            R[1024 + M * 64 + c] = (_Float16)v;
            int dst = (e0 + M) / KNB;
            atomicAdd(&m_i[(size_t)dst * 64 + c], v);
        }
    }
    __syncthreads();

    // ---- GEMM3: m @ coord_w1 -> ch (silu)
    zero_acc4(acc);
    {
        const _Float16* Ab = R + 1024 + m * 64;
#pragma unroll
        for (int ks = 0; ks < 2; ++ks) {
            v16h a = load_a_frag(Ab, ks * 32, half);
#pragma unroll
            for (int nt = 0; nt < 4; ++nt) {
                v16h b = *(const v16h*)(pcw1 + ((size_t)(ks * 4 + nt) * 32 + lane) * 16);
                acc[nt] = wmma_f16f32(a, b, acc[nt]);
            }
        }
    }
    __syncthreads();
#pragma unroll
    for (int nt = 0; nt < 4; ++nt) {
        int c = nt * 16 + m;
        float bi = cb1[c];
#pragma unroll
        for (int r = 0; r < 8; ++r) {
            float v = silu_f(acc[nt][r] + bi);
            R[2048 + (r + 8 * half) * 64 + c] = (_Float16)v;
        }
    }
    __syncthreads();

    // ---- gate = ch @ coord_w2 ; coord_shift = diff/dist * gate (atomic)
    if (lane < 16) {
        int e = e0 + lane;
        int src = edge_src[e], dst = e / KNB;
        const v8h* chv = (const v8h*)(R + 2048 + lane * 64);
        float g = 0.0f;
#pragma unroll
        for (int jj = 0; jj < 8; ++jj) {
            v8h c8 = chv[jj];
            float4 w0 = *(const float4*)(cw2 + jj * 8);
            float4 w1 = *(const float4*)(cw2 + jj * 8 + 4);
            g += (float)c8[0] * w0.x + (float)c8[1] * w0.y +
                 (float)c8[2] * w0.z + (float)c8[3] * w0.w +
                 (float)c8[4] * w1.x + (float)c8[5] * w1.y +
                 (float)c8[6] * w1.z + (float)c8[7] * w1.w;
        }
        float dx = cart[src * 3 + 0] - cart[dst * 3 + 0];
        float dy = cart[src * 3 + 1] - cart[dst * 3 + 1];
        float dz = cart[src * 3 + 2] - cart[dst * 3 + 2];
        float dsq = dx * dx + dy * dy + dz * dz;
        float inv = g / sqrtf(dsq + 1e-8f);
        atomicAdd(&coord_out[dst * 3 + 0], dx * inv);
        atomicAdd(&coord_out[dst * 3 + 1], dy * inv);
        atomicAdd(&coord_out[dst * 3 + 2], dz * inv);
    }
}

// ---------------------------------------------------------------------------
// Node update: h += MLP([h | m_i | t_nodes]). grid=64, block=256.
// LDS per wave: 3072 halves. [0,3072): n_in (16x192); hid1 reuses [0,1024).
// If out_h != nullptr, also writes the updated h there (final layer).
// ---------------------------------------------------------------------------
__global__ void node_kernel(float* __restrict__ h, const float* __restrict__ m_i,
                            const float* __restrict__ te, const int* __restrict__ batch,
                            const _Float16* __restrict__ pn1, const float* __restrict__ nb1,
                            const _Float16* __restrict__ pn2, const float* __restrict__ nb2,
                            float* __restrict__ out_h) {
    __shared__ alignas(32) _Float16 sm[8 * 3072];
    int lane = threadIdx.x & 31, wave = threadIdx.x >> 5;
    int r0 = (blockIdx.x * 8 + wave) * 16;
    _Float16* R = sm + (size_t)wave * 3072;
    {
        int row = lane >> 1, sub = lane & 1;
        int node = r0 + row;
        const float* hr = h + (size_t)node * 64 + sub * 32;
        const float* mr = m_i + (size_t)node * 64 + sub * 32;
        const float* tr = te + (size_t)batch[node] * 64 + sub * 32;
        _Float16* out = R + row * 192 + sub * 32;
#pragma unroll
        for (int j = 0; j < 8; ++j)
            *(v4h*)(out + j * 4) = cvt4(*(const float4*)(hr + j * 4));
#pragma unroll
        for (int j = 0; j < 8; ++j)
            *(v4h*)(out + 64 + j * 4) = cvt4(*(const float4*)(mr + j * 4));
#pragma unroll
        for (int j = 0; j < 8; ++j)
            *(v4h*)(out + 128 + j * 4) = cvt4(*(const float4*)(tr + j * 4));
    }
    __syncthreads();

    int m = lane & 15, half = lane >> 4;
    v8f acc[4];

    zero_acc4(acc);
    {
        const _Float16* Ab = R + m * 192;
#pragma unroll
        for (int ks = 0; ks < 6; ++ks) {
            v16h a = load_a_frag(Ab, ks * 32, half);
#pragma unroll
            for (int nt = 0; nt < 4; ++nt) {
                v16h b = *(const v16h*)(pn1 + ((size_t)(ks * 4 + nt) * 32 + lane) * 16);
                acc[nt] = wmma_f16f32(a, b, acc[nt]);
            }
        }
    }
    __syncthreads();
#pragma unroll
    for (int nt = 0; nt < 4; ++nt) {
        int c = nt * 16 + m;
        float bi = nb1[c];
#pragma unroll
        for (int r = 0; r < 8; ++r) {
            float v = silu_f(acc[nt][r] + bi);
            R[(r + 8 * half) * 64 + c] = (_Float16)v;
        }
    }
    __syncthreads();

    zero_acc4(acc);
    {
        const _Float16* Ab = R + m * 64;
#pragma unroll
        for (int ks = 0; ks < 2; ++ks) {
            v16h a = load_a_frag(Ab, ks * 32, half);
#pragma unroll
            for (int nt = 0; nt < 4; ++nt) {
                v16h b = *(const v16h*)(pn2 + ((size_t)(ks * 4 + nt) * 32 + lane) * 16);
                acc[nt] = wmma_f16f32(a, b, acc[nt]);
            }
        }
    }
#pragma unroll
    for (int nt = 0; nt < 4; ++nt) {
        int c = nt * 16 + m;
        float bi = nb2[c];
#pragma unroll
        for (int r = 0; r < 8; ++r) {
            int row = r0 + r + 8 * half;
            size_t idx = (size_t)row * 64 + c;
            float nv = h[idx] + acc[nt][r] + bi;
            h[idx] = nv;
            if (out_h) out_h[idx] = nv;
        }
    }
}

// ---------------------------------------------------------------------------
extern "C" void kernel_launch(void* const* d_in, const int* in_sizes, int n_in,
                              void* d_out, int out_size, void* d_ws, size_t ws_size,
                              hipStream_t stream) {
    (void)in_sizes; (void)out_size; (void)ws_size;
    // Inputs (JAX sorted-key pytree flatten assumed):
    const float* z     = (const float*)d_in[0];
    const float* tarr  = (const float*)d_in[1];
    const float* cart  = (const float*)d_in[2];
    const int*   batch = (const int*)d_in[3];
    const float* emb_b = (const float*)d_in[4];
    const float* emb_w = (const float*)d_in[5];
    // layers: base 6 + 11*l:
    //  0 coord_b1, 1 coord_w1, 2 coord_w2, 3 edge_b1, 4 edge_b2,
    //  5 edge_w1, 6 edge_w2, 7 node_b1, 8 node_b2, 9 node_w1, 10 node_w2
    const float* time_b1 = (const float*)d_in[50];
    const float* time_b2 = (const float*)d_in[51];
    const float* time_w1 = (const float*)d_in[52];
    const float* time_w2 = (const float*)d_in[53];
    if (n_in < 54) return;

    char* ws = (char*)d_ws;
    size_t cur = 0;
    auto take = [&](size_t bytes) -> void* {
        void* p = ws + cur;
        cur += (bytes + 255) & ~(size_t)255;
        return p;
    };
    float* te   = (float*)take((size_t)BB * TIMEDIM * 4);
    float* hbuf = (float*)take((size_t)NN * NODE * 4);
    float* mi   = (float*)take((size_t)NN * NODE * 4);
    int*   esrc = (int*)take((size_t)NEDGE * 4);
    _Float16* p_emb = (_Float16*)take((size_t)4 * 4 * 32 * 16 * 2);
    _Float16 *p_ew1[NLAYER], *p_ew2[NLAYER], *p_cw1[NLAYER], *p_nw1[NLAYER], *p_nw2[NLAYER];
    for (int l = 0; l < NLAYER; ++l) {
        p_ew1[l] = (_Float16*)take((size_t)7 * 4 * 32 * 16 * 2);
        p_ew2[l] = (_Float16*)take((size_t)2 * 4 * 32 * 16 * 2);
        p_cw1[l] = (_Float16*)take((size_t)2 * 4 * 32 * 16 * 2);
        p_nw1[l] = (_Float16*)take((size_t)6 * 4 * 32 * 16 * 2);
        p_nw2[l] = (_Float16*)take((size_t)2 * 4 * 32 * 16 * 2);
    }

    float* out_coord = (float*)d_out;              // [8192*3]
    float* out_h     = (float*)d_out + NN * 3;     // [8192*64]

    zero_f32_kernel<<<(NN * 3 + 255) / 256, 256, 0, stream>>>(out_coord, NN * 3);
    time_mlp_kernel<<<BB, 64, 0, stream>>>(tarr, time_w1, time_b1, time_w2, time_b2, te);
    pack_weights_kernel<<<2, 256, 0, stream>>>(emb_w, p_emb, 128, 4, 0);
    for (int l = 0; l < NLAYER; ++l) {
        int base = 6 + 11 * l;
        pack_weights_kernel<<<4, 256, 0, stream>>>((const float*)d_in[base + 5], p_ew1[l], 193, 7, 1);
        pack_weights_kernel<<<1, 256, 0, stream>>>((const float*)d_in[base + 6], p_ew2[l], 64, 2, 0);
        pack_weights_kernel<<<1, 256, 0, stream>>>((const float*)d_in[base + 1], p_cw1[l], 64, 2, 0);
        pack_weights_kernel<<<3, 256, 0, stream>>>((const float*)d_in[base + 9], p_nw1[l], 192, 6, 0);
        pack_weights_kernel<<<1, 256, 0, stream>>>((const float*)d_in[base + 10], p_nw2[l], 64, 2, 0);
    }
    knn_kernel<<<BB, 128, 0, stream>>>(cart, esrc);
    embed_kernel<<<NN / 128, 256, 0, stream>>>(z, p_emb, emb_b, hbuf);

    for (int l = 0; l < NLAYER; ++l) {
        int base = 6 + 11 * l;
        zero_f32_kernel<<<(NN * NODE + 255) / 256, 256, 0, stream>>>(mi, NN * NODE);
        edge_kernel<<<NEDGE / 128, 256, 0, stream>>>(
            hbuf, te, batch, cart, esrc,
            p_ew1[l], (const float*)d_in[base + 3],
            p_ew2[l], (const float*)d_in[base + 4],
            p_cw1[l], (const float*)d_in[base + 0],
            (const float*)d_in[base + 2],
            mi, out_coord);
        node_kernel<<<NN / 128, 256, 0, stream>>>(
            hbuf, mi, te, batch,
            p_nw1[l], (const float*)d_in[base + 7],
            p_nw2[l], (const float*)d_in[base + 8],
            (l == NLAYER - 1) ? out_h : (float*)nullptr);
    }
}
